// Qwen3NextGatedDeltaNet_84361747628517
// MI455X (gfx1250) — compile-verified, matrix operand order checked by reading, not run on hardware
//
#include <hip/hip_runtime.h>
#include <math.h>

// ---------------------------------------------------------------------------
// Qwen3-Next GatedDeltaNet for gfx1250 (MI455X), wave32 + WMMA bf16 + TDM.
//   1. cvt: hidden f32->bf16; W_qkvz/W_out f32->bf16 WITH transpose (so both
//      GEMM operands are row-major along K -> TDM-stageable, no LDS transpose)
//   2. WMMA GEMM: qkvz = X @ W_qkvz  (4096x2048x12288)
//      - A and B^T tiles double-buffered in LDS via TENSOR_LOAD_TO_LDS
//        (padded rows: pad_interval=16 DW, pad_amount=4 DW -> stride 40 elems)
//   3. scalar GEMM: ba = X @ W_ba (tiny, N=64)
//   4. causal depthwise conv(K=4)+SiLU with qkvz->mixed channel remap
//   5. per-head L2 norm of q (x DK^-0.5) and k
//   6. gates: beta=sigmoid(b), alpha=exp(-exp(A_log)*softplus(a+dt_bias))
//   7. delta-rule recurrence: 64 blocks (b,h), state column per thread in regs
//   8. gated RMSNorm -> bf16
//   9. WMMA GEMM: out = og @ W_out  (4096x4096x2048)
// ---------------------------------------------------------------------------

typedef __attribute__((ext_vector_type(16))) __bf16 v16bf;
typedef __attribute__((ext_vector_type(8)))  float  v8f;
typedef __attribute__((ext_vector_type(4)))  unsigned int v4u;
typedef __attribute__((ext_vector_type(8)))  int    v8i32;
typedef __attribute__((ext_vector_type(4)))  int    v4i32;

#if defined(__has_builtin)
#if __has_builtin(__builtin_amdgcn_tensor_load_to_lds) && \
    __has_builtin(__builtin_amdgcn_s_wait_tensorcnt)
#define HAVE_TDM 1
#endif
#endif
#if __has_include(<hip/amd_detail/amd_gfx1250_TDM.h>)
#define TDM_6ARG 1
#endif

#define B_     2
#define T_     2048
#define HID_   2048
#define NKH_   16
#define NVH_   32
#define DK_    128
#define DV_    128
#define KDIM_  2048
#define VDIM_  4096
#define QKVZ_N (2*KDIM_ + 2*VDIM_)   // 12288
#define HS_    (2*DK_ + 2*2*DV_)     // 768 columns per k-head in qkvz
#define CONVC  (2*KDIM_ + VDIM_)     // 8192 conv channels
#define MT     (B_*T_)               // 4096 rows

// ---------------------------------------------------------------- conversions
__global__ void cvt_f32_bf16(const float* __restrict__ in,
                             __bf16* __restrict__ out, size_t n) {
  size_t i = (size_t)blockIdx.x * blockDim.x + threadIdx.x;
  size_t stride = (size_t)gridDim.x * blockDim.x;
  for (; i < n; i += stride) out[i] = (__bf16)in[i];
}

// Transpose-convert: in[R][C] f32 -> out[C][R] bf16 (LDS-tiled, coalesced).
__global__ void cvt_transpose_bf16(const float* __restrict__ in,
                                   __bf16* __restrict__ out, int R, int C) {
  __shared__ float tile[32][33];
  int tx = threadIdx.x, ty = threadIdx.y;
  int c0 = blockIdx.x * 32, r0 = blockIdx.y * 32;
#pragma unroll
  for (int i = 0; i < 4; ++i)
    tile[ty + i * 8][tx] = in[(size_t)(r0 + ty + i * 8) * C + c0 + tx];
  __syncthreads();
#pragma unroll
  for (int i = 0; i < 4; ++i)
    out[(size_t)(c0 + ty + i * 8) * R + r0 + tx] = (__bf16)tile[tx][ty + i * 8];
}

// ----------------------------------------------------------- TDM tile stage
#if defined(HAVE_TDM)
// Issue one Tensor-DMA: 128x32 bf16 tile (rows row0.., cols k0..) of a
// [rows x K] row-major matrix into LDS at ldsAddr with row stride 40 elems.
__device__ __forceinline__ void tdm_stage(const __bf16* base, int rows, int K,
                                          int row0, int k0, unsigned ldsAddr) {
  unsigned long long ga =
      (unsigned long long)(size_t)(base + (size_t)row0 * K + k0);
  v4u g0;
  g0[0] = 1u;                                   // count = 1 valid descriptor
  g0[1] = ldsAddr;                              // lds_addr (bytes)
  g0[2] = (unsigned)ga;                         // global_addr[31:0]
  g0[3] = ((unsigned)(ga >> 32) & 0x01FFFFFFu)  // global_addr[56:32]
          | 0x80000000u;                        // type = 2 ("image")
  v8i32 g1;
  g1[0] = (1 << 16)      // data_size = 2 bytes
        | (1 << 20)      // pad_enable
        | (3 << 22)      // pad_interval: 16 DWORDs (64B = one 32-elem row)
        | (3 << 25);     // pad_amount:   4 DWORDs (16B -> stride 40 elems)
  g1[1] = (int)(((unsigned)K & 0xFFFFu) << 16);              // tensor_dim0 lo
  g1[2] = (int)(((unsigned)K >> 16) |
                (((unsigned)rows & 0xFFFFu) << 16));         // td0 hi | td1 lo
  g1[3] = (int)(((unsigned)rows >> 16) | (32u << 16));       // td1 hi | tile_dim0
  g1[4] = 128;                                               // tile_dim1 (td2=0)
  g1[5] = K;                                                 // td0_stride lo
  g1[6] = 0;
  g1[7] = 0;
  v4i32 gz = {0, 0, 0, 0};
#if defined(TDM_6ARG)
  v8i32 gz8 = {0, 0, 0, 0, 0, 0, 0, 0};
  __builtin_amdgcn_tensor_load_to_lds(g0, g1, gz, gz, gz8, 0);
#else
  __builtin_amdgcn_tensor_load_to_lds(g0, g1, gz, gz, 0);
#endif
}
#else
// Fallback cooperative stage (no transpose needed: src is row-major along K).
__device__ __forceinline__ void coop_stage(const __bf16* __restrict__ src,
                                           int row0, int K, int k0,
                                           __bf16* dst, int tid) {
  for (int g = tid; g < 512; g += 256) {
    int row = g >> 2, cg = g & 3;
    uint4 d = *(const uint4*)(src + (size_t)(row0 + row) * K + k0 + cg * 8);
    *(uint4*)(&dst[row * 40 + cg * 8]) = d;
  }
}
#endif

// ------------------------------------------------------------- WMMA bf16 GEMM
// C[M,N] = A[M,K] * Bt[N,K]^T.  Block tile 128x128, 8 waves (2x4),
// wave tile 64x32 = 4x2 wmma 16x16x32.  Double-buffered LDS tiles.
__launch_bounds__(256)
__global__ void gemm_bf16_wmma(const __bf16* __restrict__ A,
                               const __bf16* __restrict__ Bt,
                               float* __restrict__ C, int M, int N, int K) {
  __shared__ __align__(16) __bf16 lA[2][128 * 40];  // [row][k], stride 40
  __shared__ __align__(16) __bf16 lB[2][128 * 40];  // [n][k],   stride 40
  const int tid  = threadIdx.x;
  const int lane = tid & 31;
  const int wave = tid >> 5;
  const int wm   = wave >> 2;       // 0..1
  const int wn   = wave & 3;        // 0..3
  const int bM   = blockIdx.y * 128;
  const int bN   = blockIdx.x * 128;

  const v8f vz = {0.f, 0.f, 0.f, 0.f, 0.f, 0.f, 0.f, 0.f};
  v8f acc[4][2];
  for (int i = 0; i < 4; ++i)
    for (int j = 0; j < 2; ++j) acc[i][j] = vz;

  const int rrow = lane & 15;
  const int kb   = (lane >> 4) * 8;   // K base for this half-wave (ISA A layout)
  const int nsteps = K >> 5;

  // prologue: fill buffer 0
#if defined(HAVE_TDM)
  if (wave == 0)      tdm_stage(A,  M, K, bM, 0, (unsigned)(size_t)&lA[0][0]);
  else if (wave == 1) tdm_stage(Bt, N, K, bN, 0, (unsigned)(size_t)&lB[0][0]);
  if (wave < 2) __builtin_amdgcn_s_wait_tensorcnt(0);
#else
  coop_stage(A,  bM, K, 0, &lA[0][0], tid);
  coop_stage(Bt, bN, K, 0, &lB[0][0], tid);
#endif
  __syncthreads();

  int cur = 0;
  for (int i = 0; i < nsteps; ++i) {
    const int k0n = (i + 1) << 5;
    // kick off next tiles into the other buffer (async overlap with compute)
    if (i + 1 < nsteps) {
#if defined(HAVE_TDM)
      if (wave == 0)
        tdm_stage(A,  M, K, bM, k0n, (unsigned)(size_t)&lA[cur ^ 1][0]);
      else if (wave == 1)
        tdm_stage(Bt, N, K, bN, k0n, (unsigned)(size_t)&lB[cur ^ 1][0]);
#endif
      if (i + 2 < nsteps) {  // global_prefetch_b8 of the tile after next
        __builtin_prefetch(A  + (size_t)(bM + (tid >> 1)) * K + k0n + 32, 0, 1);
        __builtin_prefetch(Bt + (size_t)(bN + (tid >> 1)) * K + k0n + 32, 0, 1);
      }
    }

    // fragments per ISA 16-bit A layout: elems 0..7 -> K=kb..kb+7,
    // elems 8..15 -> K=kb+16..kb+23 (two 16B LDS reads each)
    v16bf af[4], bfr[2];
#pragma unroll
    for (int x = 0; x < 4; ++x) {
      const __bf16* p = &lA[cur][(wm * 64 + x * 16 + rrow) * 40 + kb];
      ((uint4*)&af[x])[0] = *(const uint4*)p;
      ((uint4*)&af[x])[1] = *(const uint4*)(p + 16);
    }
#pragma unroll
    for (int y = 0; y < 2; ++y) {
      const __bf16* p = &lB[cur][(wn * 32 + y * 16 + rrow) * 40 + kb];
      ((uint4*)&bfr[y])[0] = *(const uint4*)p;
      ((uint4*)&bfr[y])[1] = *(const uint4*)(p + 16);
    }
#pragma unroll
    for (int x = 0; x < 4; ++x)
#pragma unroll
      for (int y = 0; y < 2; ++y)
        acc[x][y] = __builtin_amdgcn_wmma_f32_16x16x32_bf16(
            false, af[x], false, bfr[y], (short)0, acc[x][y], false, false);

#if !defined(HAVE_TDM)
    if (i + 1 < nsteps) {
      coop_stage(A,  bM, K, k0n, &lA[cur ^ 1][0], tid);
      coop_stage(Bt, bN, K, k0n, &lB[cur ^ 1][0], tid);
    }
#endif
#if defined(HAVE_TDM)
    if (i + 1 < nsteps && wave < 2) __builtin_amdgcn_s_wait_tensorcnt(0);
#endif
    __syncthreads();
    cur ^= 1;
  }

  // D layout: lanes 0-15 -> M=r, lanes 16-31 -> M=8+r; N = lane%16
  const int mo = (lane >> 4) * 8;
  const int no = lane & 15;
#pragma unroll
  for (int x = 0; x < 4; ++x)
#pragma unroll
    for (int y = 0; y < 2; ++y) {
      int mBase = bM + wm * 64 + x * 16 + mo;
      int n     = bN + wn * 32 + y * 16 + no;
#pragma unroll
      for (int r = 0; r < 8; ++r)
        C[(size_t)(mBase + r) * N + n] = acc[x][y][r];
    }
}

// ----------------------------------------------------------------- ba (small)
__global__ void ba_gemm(const float* __restrict__ h, const float* __restrict__ W,
                        float* __restrict__ out) {
  int id = blockIdx.x * blockDim.x + threadIdx.x;
  if (id >= MT * 64) return;
  int n = id & 63, m = id >> 6;
  const float* hr = h + (size_t)m * HID_;
  float s = 0.f;
  for (int k = 0; k < HID_; ++k) s += hr[k] * W[(size_t)k * 64 + n];
  out[id] = s;
}

// ------------------------------------------------- conv(K=4)+SiLU with remap
__global__ void conv_silu(const float* __restrict__ qkvz,
                          const float* __restrict__ cw,
                          float* __restrict__ qb, float* __restrict__ kbuf,
                          float* __restrict__ vb) {
  size_t id = (size_t)blockIdx.x * blockDim.x + threadIdx.x;
  if (id >= (size_t)MT * CONVC) return;
  int c = (int)(id & (CONVC - 1));
  size_t bt = id >> 13;               // CONVC == 8192
  int t = (int)(bt & (T_ - 1));
  int col, cls, h, off, vh = 0;
  if (c < KDIM_) {                    // q channels
    h = c >> 7; off = c & 127; col = h * HS_ + off; cls = 0;
  } else if (c < 2 * KDIM_) {         // k channels
    int c2 = c - KDIM_; h = c2 >> 7; off = c2 & 127;
    col = h * HS_ + DK_ + off; cls = 1;
  } else {                            // v channels
    int c3 = c - 2 * KDIM_; vh = c3 >> 7; off = c3 & 127;
    h = vh >> 1; int r = vh & 1;
    col = h * HS_ + 2 * DK_ + r * DV_ + off; cls = 2;
  }
  float s = 0.f;
#pragma unroll
  for (int j = 0; j < 4; ++j) {
    int tau = t - 3 + j;
    if (tau >= 0) s += qkvz[(bt - t + tau) * QKVZ_N + col] * cw[c * 4 + j];
  }
  s = s / (1.f + __expf(-s));         // SiLU
  if (cls == 0)      qb[(bt * NKH_ + h) * DK_ + off] = s;
  else if (cls == 1) kbuf[(bt * NKH_ + h) * DK_ + off] = s;
  else               vb[(bt * NVH_ + vh) * DV_ + off] = s;
}

// ------------------------------------------------------------- q/k L2 norm
__global__ void l2norm_qk(float* __restrict__ qb, float* __restrict__ kbuf) {
  __shared__ float sm[DK_];
  int tid = threadIdx.x;
  size_t base = (size_t)blockIdx.x * DK_;   // grid = MT*NKH
  float q = qb[base + tid];
  sm[tid] = q * q; __syncthreads();
  for (int s = 64; s > 0; s >>= 1) { if (tid < s) sm[tid] += sm[tid + s]; __syncthreads(); }
  float invq = rsqrtf(sm[0] + 1e-6f) * 0.08838834764831845f;  // * DK^-0.5
  __syncthreads();
  qb[base + tid] = q * invq;
  float k = kbuf[base + tid];
  sm[tid] = k * k; __syncthreads();
  for (int s = 64; s > 0; s >>= 1) { if (tid < s) sm[tid] += sm[tid + s]; __syncthreads(); }
  kbuf[base + tid] = k * rsqrtf(sm[0] + 1e-6f);
}

// --------------------------------------------------------------------- gates
__global__ void gates_kernel(const float* __restrict__ ba,
                             const float* __restrict__ dt_bias,
                             const float* __restrict__ A_log,
                             float* __restrict__ alpha, float* __restrict__ beta) {
  int id = blockIdx.x * blockDim.x + threadIdx.x;
  if (id >= MT * NVH_) return;
  int vh = id & 31; size_t bt = (size_t)id >> 5;
  int kh = vh >> 1, r = vh & 1;
  float b = ba[bt * 64 + kh * 4 + r];
  float a = ba[bt * 64 + kh * 4 + 2 + r];
  float x = a + dt_bias[vh];
  float sp = (x > 20.f) ? x : log1pf(__expf(x));
  alpha[id] = __expf(-__expf(A_log[vh]) * sp);
  beta[id]  = 1.f / (1.f + __expf(-b));
}

// ------------------------------------------------- gated delta-rule recurrence
// One block per (b, vh); thread tid owns state column S[:, tid] in registers.
__launch_bounds__(128)
__global__ void delta_recurrence(const float* __restrict__ qb,
                                 const float* __restrict__ kbuf,
                                 const float* __restrict__ vb,
                                 const float* __restrict__ alpha,
                                 const float* __restrict__ beta,
                                 float* __restrict__ ob) {
  __shared__ float qs[DK_], ks[DK_];
  const int tid = threadIdx.x;
  const int b   = blockIdx.x >> 5;
  const int vh  = blockIdx.x & 31;
  const int kh  = vh >> 1;
  float S[DK_];
#pragma unroll
  for (int k = 0; k < DK_; ++k) S[k] = 0.f;
  for (int t = 0; t < T_; ++t) {
    size_t bt = (size_t)b * T_ + t;
    qs[tid] = qb[(bt * NKH_ + kh) * DK_ + tid];
    ks[tid] = kbuf[(bt * NKH_ + kh) * DK_ + tid];
    float vv = vb[(bt * NVH_ + vh) * DV_ + tid];
    float al = alpha[bt * NVH_ + vh];
    float be = beta[bt * NVH_ + vh];
    __syncthreads();
    float err = 0.f;
#pragma unroll
    for (int k = 0; k < DK_; ++k) { float s = S[k] * al; S[k] = s; err += s * ks[k]; }
    float vnew = (vv - err) * be;
    float oa = 0.f;
#pragma unroll
    for (int k = 0; k < DK_; ++k) { float s = S[k] + ks[k] * vnew; S[k] = s; oa += s * qs[k]; }
    ob[(bt * NVH_ + vh) * DV_ + tid] = oa;
    __syncthreads();
  }
}

// -------------------------------------------------- gated RMSNorm -> bf16
__global__ void gated_rmsnorm(const float* __restrict__ ob,
                              const float* __restrict__ qkvz,
                              const float* __restrict__ nw,
                              __bf16* __restrict__ og) {
  __shared__ float sm[DV_];
  int tid = threadIdx.x;
  int idx = blockIdx.x;               // bt*NVH + vh
  int vh = idx & 31; size_t bt = (size_t)idx >> 5;
  int kh = vh >> 1, r = vh & 1;
  float o = ob[(size_t)idx * DV_ + tid];
  float z = qkvz[bt * QKVZ_N + kh * HS_ + 2 * DK_ + 2 * DV_ + r * DV_ + tid];
  float g = o * (z / (1.f + __expf(-z)));
  sm[tid] = g * g; __syncthreads();
  for (int s = 64; s > 0; s >>= 1) { if (tid < s) sm[tid] += sm[tid + s]; __syncthreads(); }
  float inv = rsqrtf(sm[0] * (1.f / (float)DV_) + 1e-6f);
  og[bt * VDIM_ + vh * DV_ + tid] = (__bf16)(g * inv * nw[tid]);
}

// ---------------------------------------------------------------------------
extern "C" void kernel_launch(void* const* d_in, const int* in_sizes, int n_in,
                              void* d_out, int out_size, void* d_ws, size_t ws_size,
                              hipStream_t stream) {
  (void)in_sizes; (void)n_in; (void)out_size; (void)ws_size;
  const float* hidden  = (const float*)d_in[0];
  const float* W_qkvz  = (const float*)d_in[1];
  const float* W_ba    = (const float*)d_in[2];
  const float* conv_w  = (const float*)d_in[3];
  const float* dt_bias = (const float*)d_in[4];
  const float* A_log   = (const float*)d_in[5];
  const float* norm_w  = (const float*)d_in[6];
  const float* W_out   = (const float*)d_in[7];
  float* out = (float*)d_out;

  char* ws = (char*)d_ws;
  size_t off = 0;
  auto take = [&](size_t bytes) {
    char* p = ws + off;
    off += (bytes + 255) & ~(size_t)255;
    return p;
  };
  __bf16* hiddenB = (__bf16*)take((size_t)MT * HID_ * 2);
  __bf16* WqkvzT  = (__bf16*)take((size_t)HID_ * QKVZ_N * 2);  // [12288][2048]
  float*  qkvz    = (float*) take((size_t)MT * QKVZ_N * 4);
  float*  ba      = (float*) take((size_t)MT * 64 * 4);
  float*  qb      = (float*) take((size_t)MT * KDIM_ * 4);
  float*  kb      = (float*) take((size_t)MT * KDIM_ * 4);
  float*  vb      = (float*) take((size_t)MT * VDIM_ * 4);
  float*  alpha   = (float*) take((size_t)MT * NVH_ * 4);
  float*  beta    = (float*) take((size_t)MT * NVH_ * 4);
  float*  ob      = (float*) take((size_t)MT * VDIM_ * 4);
  // Reuse WqkvzT region (50.3 MB) for og bf16 (33.5 MB) + W_out^T bf16
  // (16.8 MB): both only live after GEMM1 has consumed WqkvzT.
  __bf16* ogB   = (__bf16*)WqkvzT;
  __bf16* WoutT = (__bf16*)((char*)WqkvzT + (size_t)MT * VDIM_ * 2); // [2048][4096]

  // 1. conversions (weights transposed so GEMM B-operand is row-major on K)
  cvt_f32_bf16<<<4096, 256, 0, stream>>>(hidden, hiddenB, (size_t)MT * HID_);
  cvt_transpose_bf16<<<dim3(QKVZ_N / 32, HID_ / 32), dim3(32, 8), 0, stream>>>(
      W_qkvz, WqkvzT, HID_, QKVZ_N);

  // 2. qkvz = X @ W_qkvz  (WMMA bf16, TDM double-buffered tiles)
  gemm_bf16_wmma<<<dim3(QKVZ_N / 128, MT / 128), 256, 0, stream>>>(
      hiddenB, WqkvzT, qkvz, MT, QKVZ_N, HID_);

  // 3. ba = X @ W_ba (tiny)
  ba_gemm<<<(MT * 64) / 256, 256, 0, stream>>>(hidden, W_ba, ba);

  // 4. conv + SiLU
  conv_silu<<<(unsigned)(((size_t)MT * CONVC) / 256), 256, 0, stream>>>(
      qkvz, conv_w, qb, kb, vb);

  // 5. q/k L2 norm
  l2norm_qk<<<MT * NKH_, 128, 0, stream>>>(qb, kb);

  // 6. gates
  gates_kernel<<<(MT * NVH_) / 256, 256, 0, stream>>>(ba, dt_bias, A_log, alpha, beta);

  // 7. recurrence
  delta_recurrence<<<B_ * NVH_, 128, 0, stream>>>(qb, kb, vb, alpha, beta, ob);

  // 8. gated RMSNorm -> bf16 (writes into reused WqkvzT region)
  gated_rmsnorm<<<MT * NVH_, 128, 0, stream>>>(ob, qkvz, norm_w, ogB);

  // 9. out = og @ W_out (WMMA bf16)
  cvt_transpose_bf16<<<dim3(HID_ / 32, VDIM_ / 32), dim3(32, 8), 0, stream>>>(
      W_out, WoutT, VDIM_, HID_);
  gemm_bf16_wmma<<<dim3(HID_ / 128, MT / 128), 256, 0, stream>>>(
      ogB, WoutT, out, MT, HID_, VDIM_);
}
